// VoteNet_77532749627400
// MI455X (gfx1250) — matrix operand order repaired and checked by reading, not compiled, not run on hardware
//
#include <hip/hip_runtime.h>
#include <hip/hip_bf16.h>

typedef __attribute__((ext_vector_type(16))) _Float16 v16h;
typedef __attribute__((ext_vector_type(8)))  float    v8f;

#define BB 4
#define N0 8192
#define C0 16

// ---------------- transpose input (B,C,N) -> channel-major [c][b*N+n] ----------
__global__ __launch_bounds__(256) void k_transpose_in(const float* __restrict__ in,
                                                      float* __restrict__ out) {
  int n = blockIdx.x * 256 + threadIdx.x;
  int c = blockIdx.y, b = blockIdx.z;
  out[(size_t)c * (BB * N0) + (size_t)b * N0 + n] =
      in[((size_t)b * C0 + c) * N0 + n];
}

// ---------------- farthest point sampling: 1 workgroup per batch --------------
__global__ __launch_bounds__(256) void k_fps(const float* __restrict__ xyz, int cs,
                                             int N, int S,
                                             int* __restrict__ fidx,
                                             float* __restrict__ nxyz, int ncs) {
  __shared__ float dist[8192];
  __shared__ float rv[256];
  __shared__ int   ri[256];
  int b = blockIdx.x, tid = threadIdx.x;
  const float* px = xyz + (size_t)b * N;
  const float* py = xyz + (size_t)cs + (size_t)b * N;
  const float* pz = xyz + (size_t)2 * cs + (size_t)b * N;
  for (int i = tid; i < N; i += 256) dist[i] = 1e10f;
  __syncthreads();
  int far = 0;
  for (int it = 0; it < S; ++it) {
    if (tid == 0) {
      fidx[b * S + it] = far;
      nxyz[(size_t)0 * ncs + b * S + it] = px[far];
      nxyz[(size_t)1 * ncs + b * S + it] = py[far];
      nxyz[(size_t)2 * ncs + b * S + it] = pz[far];
    }
    float cx = px[far], cy = py[far], cz = pz[far];
    float bv = -1.0f; int bi = 0;
    for (int i = tid; i < N; i += 256) {
      float dx = px[i] - cx, dy = py[i] - cy, dz = pz[i] - cz;
      float d = dx * dx + dy * dy + dz * dz;
      float old = dist[i];
      d = fminf(old, d);
      dist[i] = d;
      if (d > bv) { bv = d; bi = i; }
    }
    rv[tid] = bv; ri[tid] = bi;
    __syncthreads();
    for (int off = 128; off > 0; off >>= 1) {
      if (tid < off) {
        float v2 = rv[tid + off]; int i2 = ri[tid + off];
        if (v2 > rv[tid] || (v2 == rv[tid] && i2 < ri[tid])) { rv[tid] = v2; ri[tid] = i2; }
      }
      __syncthreads();
    }
    far = ri[0];
    __syncthreads();
  }
}

// ---------------- ball query: first-K in-radius indices, pad with first -------
__global__ __launch_bounds__(256) void k_ballq(const float* __restrict__ xyz, int cs,
                                               const float* __restrict__ nxyz, int ncs,
                                               int N, int S, float r2, int K,
                                               int* __restrict__ idx) {
  int s = blockIdx.x * 256 + threadIdx.x;
  int b = blockIdx.y;
  if (s >= S) return;
  float cx = nxyz[b * S + s];
  float cy = nxyz[(size_t)ncs + b * S + s];
  float cz = nxyz[(size_t)2 * ncs + b * S + s];
  const float* px = xyz + (size_t)b * N;
  const float* py = xyz + (size_t)cs + (size_t)b * N;
  const float* pz = xyz + (size_t)2 * cs + (size_t)b * N;
  int* o = idx + ((size_t)b * S + s) * K;
  int cnt = 0, first = -1;
  for (int n = 0; n < N && cnt < K; ++n) {
    float dx = px[n] - cx, dy = py[n] - cy, dz = pz[n] - cz;
    float d = dx * dx + dy * dy + dz * dz;
    if (d <= r2) { if (first < 0) first = n; o[cnt++] = n; }
  }
  if (first < 0) first = 0;
  for (; cnt < K; ++cnt) o[cnt] = first;
}

// ---------------- grouping: build [3+Cp][ (b*K+k)*S + s ] --------------------
__global__ __launch_bounds__(256) void k_group(const float* __restrict__ xyz, int xcs,
                                               const float* __restrict__ pts, int pcs, int Cp,
                                               const float* __restrict__ nxyz, int ncs,
                                               const int* __restrict__ idx,
                                               float* __restrict__ G,
                                               int N, int S, int K) {
  int s = blockIdx.x * 256 + threadIdx.x;
  int k = blockIdx.y, b = blockIdx.z;
  if (s >= S) return;
  int j = idx[((size_t)b * S + s) * K + k];
  size_t col  = ((size_t)b * K + k) * S + s;
  size_t Mtot = (size_t)BB * K * S;
  for (int c = 0; c < 3; ++c)
    G[(size_t)c * Mtot + col] =
        xyz[(size_t)c * xcs + (size_t)b * N + j] - nxyz[(size_t)c * ncs + b * S + s];
  for (int c = 0; c < Cp; ++c)
    G[(size_t)(3 + c) * Mtot + col] = pts[(size_t)c * pcs + (size_t)b * N + j];
}

// ---------------- zero pad rows ----------------------------------------------
__global__ __launch_bounds__(256) void k_zerorows(float* __restrict__ X,
                                                  long long count) {
  long long i = (long long)blockIdx.x * 256 + threadIdx.x;
  if (i < count) X[i] = 0.0f;
}

// ---------------- pack W (O x I, f32) into WMMA A-fragment f16 layout --------
__global__ __launch_bounds__(128) void k_packw(const float* __restrict__ W,
                                               _Float16* __restrict__ WP,
                                               int O, int I, int NTO, int NTK) {
  int t = blockIdx.x * 128 + threadIdx.x;
  int total = NTO * NTK * 32;
  if (t >= total) return;
  int lane = t & 31;
  int tile = t >> 5;
  int tk = tile % NTK, to = tile / NTK;
  int m = lane & 15, kh = lane >> 4;
  int row = to * 16 + m;
  _Float16* o = WP + (size_t)t * 16;
#pragma unroll
  for (int e = 0; e < 16; ++e) {
    int jj = e >> 1, p = e & 1;
    int k  = 2 * jj + p + kh * 16;
    int ic = tk * 32 + k;
    float wv = (row < O && ic < I) ? W[(size_t)row * I + ic] : 0.0f;
    o[e] = (_Float16)wv;
  }
}

// ---------------- WMMA GEMM (O % 64 == 0, unpredicated) ----------------------
// Block = 64 output channels x 32 columns. The 4 waves cover 4 consecutive
// oc-tiles of the SAME columns, so their B loads are identical addresses
// (L0 hits) -> ~4x less L2 traffic on X. Each wave runs two WMMAs per
// A-fragment load (16x32 D strip). IPAD constexpr => unpredicated K-loop.
template <int IPAD>
__global__ __launch_bounds__(128) void k_gemm_t(const _Float16* __restrict__ WP,
                                                const float* __restrict__ bias,
                                                const float* __restrict__ X,
                                                float* __restrict__ Y,
                                                long long Mtot) {
  constexpr int NTK = IPAD / 32;
  int lane = threadIdx.x & 31;
  int wave = threadIdx.x >> 5;
  int m    = lane & 15;
  int kh   = lane >> 4;
  long long col0 = (long long)blockIdx.x * 32;
  int to = blockIdx.y * 4 + wave;
  const _Float16* wp = WP + ((size_t)to * NTK * 32 + lane) * 16;
  float bv[8];
#pragma unroll
  for (int r = 0; r < 8; ++r) bv[r] = bias[to * 16 + 8 * kh + r];
  v8f acc0 = {}, acc1 = {};
  for (int tk = 0; tk < NTK; ++tk) {
    v16h a = *(const v16h*)(wp + (size_t)tk * 512);
    const float* xp = X + (long long)(tk * 32 + kh * 16) * Mtot + col0 + m;
    __builtin_prefetch(xp + (long long)32 * Mtot, 0, 0);
    v16h b0, b1;
#pragma unroll
    for (int e = 0; e < 16; ++e) {
      b0[e] = (_Float16)xp[(long long)e * Mtot];
      b1[e] = (_Float16)xp[(long long)e * Mtot + 16];
    }
    acc0 = __builtin_amdgcn_wmma_f32_16x16x32_f16(false, a, false, b0,
                                                  (short)0, acc0, false, false);
    acc1 = __builtin_amdgcn_wmma_f32_16x16x32_f16(false, a, false, b1,
                                                  (short)0, acc1, false, false);
  }
#pragma unroll
  for (int r = 0; r < 8; ++r) {
    int oc = to * 16 + 8 * kh + r;
    float* yp = Y + (long long)oc * Mtot + col0 + m;
    yp[0]  = acc0[r] + bv[r];
    yp[16] = acc1[r] + bv[r];
  }
}

// ---------------- WMMA GEMM, vote head (O=3 hardcoded, I=256) ----------------
// Rows 3..15 of the D tile are discarded at compile time: epilogue is one
// uniform kh==0 branch with 3 store pairs (no per-row exec-mask chains).
__global__ __launch_bounds__(128) void k_gemm_vote(const _Float16* __restrict__ WP,
                                                   const float* __restrict__ bias,
                                                   const float* __restrict__ X,
                                                   float* __restrict__ Y,
                                                   long long Mtot) {
  constexpr int NTK = 8;  // IPAD = 256
  int lane = threadIdx.x & 31;
  int wave = threadIdx.x >> 5;
  int m    = lane & 15;
  int kh   = lane >> 4;
  long long col0 = ((long long)blockIdx.x * 4 + wave) * 32;
  const _Float16* wp = WP + (size_t)lane * 16;
  v8f acc0 = {}, acc1 = {};
  for (int tk = 0; tk < NTK; ++tk) {
    v16h a = *(const v16h*)(wp + (size_t)tk * 512);
    const float* xp = X + (long long)(tk * 32 + kh * 16) * Mtot + col0 + m;
    v16h b0, b1;
#pragma unroll
    for (int e = 0; e < 16; ++e) {
      b0[e] = (_Float16)xp[(long long)e * Mtot];
      b1[e] = (_Float16)xp[(long long)e * Mtot + 16];
    }
    acc0 = __builtin_amdgcn_wmma_f32_16x16x32_f16(false, a, false, b0,
                                                  (short)0, acc0, false, false);
    acc1 = __builtin_amdgcn_wmma_f32_16x16x32_f16(false, a, false, b1,
                                                  (short)0, acc1, false, false);
  }
  if (kh == 0) {  // D rows 0..7 live in lanes 0-15; only rows 0..2 are real
#pragma unroll
    for (int r = 0; r < 3; ++r) {
      float bvr = bias[r];
      float* yp = Y + (long long)r * Mtot + col0 + m;
      yp[0]  = acc0[r] + bvr;
      yp[16] = acc1[r] + bvr;
    }
  }
}

// ---------------- BN statistics: one block per channel -----------------------
__global__ __launch_bounds__(256) void k_bnstats(const float* __restrict__ Y,
                                                 float* __restrict__ st, long long Mtot) {
  __shared__ float ss[256], s2[256];
  int c = blockIdx.x, tid = threadIdx.x;
  const float* p = Y + (long long)c * Mtot;
  float a = 0.0f, b2 = 0.0f;
  for (long long i = tid; i < Mtot; i += 256) { float v = p[i]; a += v; b2 += v * v; }
  ss[tid] = a; s2[tid] = b2; __syncthreads();
  for (int off = 128; off > 0; off >>= 1) {
    if (tid < off) { ss[tid] += ss[tid + off]; s2[tid] += s2[tid + off]; }
    __syncthreads();
  }
  if (tid == 0) {
    float mean = ss[0] / (float)Mtot;
    float var  = s2[0] / (float)Mtot - mean * mean;
    st[2 * c] = mean; st[2 * c + 1] = var;
  }
}

// ---------------- BN apply + activation (0=relu, 1=mish) ---------------------
__global__ __launch_bounds__(256) void k_bnact(float* __restrict__ Y,
                                               const float* __restrict__ gamma,
                                               const float* __restrict__ beta,
                                               const float* __restrict__ st,
                                               long long Mtot, int mode) {
  int c = blockIdx.y;
  long long i = (long long)blockIdx.x * 256 + threadIdx.x;
  if (i >= Mtot) return;
  float mean = st[2 * c], var = st[2 * c + 1];
  float v = Y[(long long)c * Mtot + i];
  v = gamma[c] * (v - mean) * rsqrtf(var + 1e-5f) + beta[c];
  if (mode == 0) {
    v = fmaxf(v, 0.0f);
  } else {
    float sp = (v > 20.0f) ? v : log1pf(expf(v));
    v = v * tanhf(sp);
  }
  Y[(long long)c * Mtot + i] = v;
}

// ---------------- max over K samples -----------------------------------------
__global__ __launch_bounds__(256) void k_maxpool(const float* __restrict__ Yin,
                                                 float* __restrict__ P,
                                                 int S, int K) {
  int s = blockIdx.x * 256 + threadIdx.x;
  int c = blockIdx.y, b = blockIdx.z;
  if (s >= S) return;
  long long Mtot = (long long)BB * K * S;
  const float* p = Yin + (long long)c * Mtot + (long long)b * K * S + s;
  float m = p[0];
  for (int k = 1; k < K; ++k) m = fmaxf(m, p[(long long)k * S]);
  P[(long long)c * (BB * S) + (long long)b * S + s] = m;
}

// ---------------- 3-NN inverse-distance interpolation into CAT[C1..C1+C2) ----
__global__ __launch_bounds__(256) void k_interp(const float* __restrict__ x1, int cs1, int N1,
                                                const float* __restrict__ x2, int cs2, int N2,
                                                const float* __restrict__ F2, int csf, int C2,
                                                float* __restrict__ CAT, int C1) {
  long long gid = (long long)blockIdx.x * 256 + threadIdx.x;
  if (gid >= (long long)BB * N1) return;
  int b = (int)(gid / N1), n = (int)(gid % N1);
  float ax = x1[b * N1 + n];
  float ay = x1[(size_t)cs1 + b * N1 + n];
  float az = x1[(size_t)2 * cs1 + b * N1 + n];
  float d0 = 1e30f, d1 = 1e30f, d2 = 1e30f; int i0 = 0, i1 = 0, i2 = 0;
  for (int j = 0; j < N2; ++j) {
    float dx = x2[b * N2 + j] - ax;
    float dy = x2[(size_t)cs2 + b * N2 + j] - ay;
    float dz = x2[(size_t)2 * cs2 + b * N2 + j] - az;
    float d = dx * dx + dy * dy + dz * dz;
    if (d < d0)      { d2 = d1; i2 = i1; d1 = d0; i1 = i0; d0 = d; i0 = j; }
    else if (d < d1) { d2 = d1; i2 = i1; d1 = d;  i1 = j; }
    else if (d < d2) { d2 = d;  i2 = j; }
  }
  float w0 = 1.0f / (d0 + 1e-8f), w1 = 1.0f / (d1 + 1e-8f), w2 = 1.0f / (d2 + 1e-8f);
  float wsum = w0 + w1 + w2; w0 /= wsum; w1 /= wsum; w2 /= wsum;
  long long outM = (long long)BB * N1;
  long long col  = (long long)b * N1 + n;
  for (int c = 0; c < C2; ++c) {
    const float* f = F2 + (long long)c * csf + (long long)b * N2;
    CAT[(long long)(C1 + c) * outM + col] = w0 * f[i0] + w1 * f[i1] + w2 * f[i2];
  }
}

// ---------------- plain channel copy (concat lower rows) ---------------------
__global__ __launch_bounds__(256) void k_copych(const float* __restrict__ src,
                                                float* __restrict__ dst,
                                                long long total) {
  long long i = (long long)blockIdx.x * 256 + threadIdx.x;
  if (i < total) dst[i] = src[i];
}

// ---------------- pack outputs: vote x, l1_xyz, fps_idx ----------------------
__global__ __launch_bounds__(256) void k_output(const float* __restrict__ vote,
                                                const float* __restrict__ nx1,
                                                const int* __restrict__ fidx,
                                                float* __restrict__ out) {
  const int S = 2048;
  int i = blockIdx.x * 256 + threadIdx.x;
  if (i >= BB * S) return;
  int b = i / S, s = i % S;
  for (int c = 0; c < 3; ++c) {
    out[((long long)b * 3 + c) * S + s] = vote[(long long)c * (BB * S) + i];
    out[(long long)3 * BB * S + ((long long)b * 3 + c) * S + s] =
        nx1[(long long)c * (BB * S) + i];
  }
  out[(long long)6 * BB * S + i] = (float)fidx[i];
}

// =============================================================================
struct Blk { const float *w[3], *b[3], *g[3], *be[3]; };

extern "C" void kernel_launch(void* const* d_in, const int* in_sizes, int n_in,
                              void* d_out, int out_size, void* d_ws, size_t ws_size,
                              hipStream_t stream) {
  (void)out_size; (void)ws_size;
  // Locate xyz (unique 4*16*8192 buffer) -> detect flatten order.
  int xi = n_in - 1;
  for (int i = 0; i < n_in; ++i) if (in_sizes[i] == BB * C0 * N0) { xi = i; break; }
  bool srt = (xi == n_in - 1);  // jax tree_flatten sorted-key order
  auto gp = [&](int i) { return (const float*)d_in[i]; };
  auto fill2 = [&](Blk& K, int t) {
    for (int l = 0; l < 2; ++l) {
      if (srt) { K.b[l] = gp(t + l); K.be[l] = gp(t + 2 + l); K.g[l] = gp(t + 4 + l); K.w[l] = gp(t + 6 + l); }
      else     { K.w[l] = gp(t + l); K.b[l]  = gp(t + 2 + l); K.g[l] = gp(t + 4 + l); K.be[l] = gp(t + 6 + l); }
    }
  };
  auto fill3 = [&](Blk& K, int t) {
    for (int l = 0; l < 3; ++l) {
      if (srt) { K.b[l] = gp(t + l); K.be[l] = gp(t + 3 + l); K.g[l] = gp(t + 6 + l); K.w[l] = gp(t + 9 + l); }
      else     { K.w[l] = gp(t + l); K.b[l]  = gp(t + 3 + l); K.g[l] = gp(t + 6 + l); K.be[l] = gp(t + 9 + l); }
    }
  };
  Blk b_sa1, b_sa2, b_sa3, b_sa4, b_fp4, b_fp3, b_fp2, b_vote;
  if (srt) {
    fill2(b_fp2, 0);  fill2(b_fp3, 8);  fill2(b_fp4, 16);
    fill3(b_sa1, 24); fill3(b_sa2, 36); fill3(b_sa3, 48); fill3(b_sa4, 60);
    fill3(b_vote, 72);
  } else {
    fill3(b_sa1, 1);  fill3(b_sa2, 13); fill3(b_sa3, 25); fill3(b_sa4, 37);
    fill2(b_fp4, 49); fill2(b_fp3, 57); fill2(b_fp2, 65); fill3(b_vote, 73);
  }
  const float* xin = gp(xi);

  // ---- workspace layout (floats) ----
  float* ws = (float*)d_ws;
  size_t off = 0;
  auto A = [&](size_t n) { float* p = ws + off; off += n; return p; };
  float* T0   = A(524288);
  float* BUFA = A(67108864ULL);
  float* BUFB = A(67108864ULL);
  float* nx1 = A(24576); float* nx2 = A(12288); float* nx3 = A(6144); float* nx4 = A(3072);
  float* P1 = A(1048576); float* P2 = A(1048576); float* P3 = A(524288); float* P4 = A(262144);
  float* L3P = A(524288); float* L2P = A(1048576); float* L1P = A(2097152);
  float* CAT4 = A(1048576); float* CAT3 = A(2097152); float* CAT2 = A(3145728);
  float* stats = A(512);
  _Float16* WPACK = (_Float16*)A(65536);  // 256x512 f16 packed-weight max
  int* fps1 = (int*)A(8192); int* fps2 = (int*)A(4096);
  int* fps3 = (int*)A(2048); int* fps4 = (int*)A(1024);
  int* bi1 = (int*)A(524288); int* bi2 = (int*)A(131072);
  int* bi3 = (int*)A(32768);  int* bi4 = (int*)A(16384);

  auto gemm = [&](const float* W, const float* bias, const float* X, float* Y,
                  int O, int I, long long M) {
    int IPAD = (I + 31) & ~31;
    int NTO = (O + 15) / 16, NTK = IPAD / 32;
    int total = NTO * NTK * 32;
    k_packw<<<dim3((total + 127) / 128), dim3(128), 0, stream>>>(W, WPACK, O, I, NTO, NTK);
    if ((O & 63) != 0) {
      // vote head: O=3, I=256
      dim3 g((unsigned)(M / 128));
      k_gemm_vote<<<g, dim3(128), 0, stream>>>(WPACK, bias, X, Y, M);
      return;
    }
    dim3 g((unsigned)(M / 32), (unsigned)(O / 64));
    switch (IPAD) {
      case 32:  k_gemm_t<32> <<<g, dim3(128), 0, stream>>>(WPACK, bias, X, Y, M); break;
      case 64:  k_gemm_t<64> <<<g, dim3(128), 0, stream>>>(WPACK, bias, X, Y, M); break;
      case 128: k_gemm_t<128><<<g, dim3(128), 0, stream>>>(WPACK, bias, X, Y, M); break;
      case 160: k_gemm_t<160><<<g, dim3(128), 0, stream>>>(WPACK, bias, X, Y, M); break;
      case 256: k_gemm_t<256><<<g, dim3(128), 0, stream>>>(WPACK, bias, X, Y, M); break;
      case 288: k_gemm_t<288><<<g, dim3(128), 0, stream>>>(WPACK, bias, X, Y, M); break;
      case 384: k_gemm_t<384><<<g, dim3(128), 0, stream>>>(WPACK, bias, X, Y, M); break;
      default:  k_gemm_t<512><<<g, dim3(128), 0, stream>>>(WPACK, bias, X, Y, M); break;
    }
  };
  auto bnact = [&](float* Y, const float* g, const float* be, int C, long long M, int mode) {
    k_bnstats<<<dim3(C), dim3(256), 0, stream>>>(Y, stats, M);
    dim3 gr((unsigned)((M + 255) / 256), (unsigned)C);
    k_bnact<<<gr, dim3(256), 0, stream>>>(Y, g, be, stats, M, mode);
  };
  auto sa = [&](const float* xyzp, int Nn, const float* pts, int Cp,
                int S, int K, float radius, const Blk& bk,
                int c0, int c1, int c2,
                int* fidx, float* nxyz, float* P, int* ball) {
    int cs = BB * Nn, ncs = BB * S;
    k_fps<<<dim3(BB), dim3(256), 0, stream>>>(xyzp, cs, Nn, S, fidx, nxyz, ncs);
    dim3 gb((S + 255) / 256, BB);
    k_ballq<<<gb, dim3(256), 0, stream>>>(xyzp, cs, nxyz, ncs, Nn, S,
                                          radius * radius, K, ball);
    dim3 gg((S + 255) / 256, K, BB);
    k_group<<<gg, dim3(256), 0, stream>>>(xyzp, cs, pts, cs, Cp, nxyz, ncs,
                                          ball, BUFA, Nn, S, K);
    long long M = (long long)BB * K * S;
    int Cin = 3 + Cp, IPAD = (Cin + 31) & ~31;
    if (IPAD > Cin) {
      long long cnt = (long long)(IPAD - Cin) * M;
      k_zerorows<<<dim3((unsigned)((cnt + 255) / 256)), dim3(256), 0, stream>>>(
          BUFA + (long long)Cin * M, cnt);
    }
    gemm(bk.w[0], bk.b[0], BUFA, BUFB, c0, Cin, M); bnact(BUFB, bk.g[0], bk.be[0], c0, M, 0);
    gemm(bk.w[1], bk.b[1], BUFB, BUFA, c1, c0, M);  bnact(BUFA, bk.g[1], bk.be[1], c1, M, 0);
    gemm(bk.w[2], bk.b[2], BUFA, BUFB, c2, c1, M);  bnact(BUFB, bk.g[2], bk.be[2], c2, M, 0);
    dim3 gm((S + 255) / 256, c2, BB);
    k_maxpool<<<gm, dim3(256), 0, stream>>>(BUFB, P, S, K);
  };
  auto fp = [&](const float* x1p, int N1, const float* x2p, int N2,
                const float* p1, int C1, const float* p2, int C2,
                const Blk& bk, float* CAT, float* OUT) {
    long long M = (long long)BB * N1;
    long long tot = (long long)C1 * M;
    k_copych<<<dim3((unsigned)((tot + 255) / 256)), dim3(256), 0, stream>>>(p1, CAT, tot);
    k_interp<<<dim3((unsigned)((M + 255) / 256)), dim3(256), 0, stream>>>(
        x1p, BB * N1, N1, x2p, BB * N2, N2, p2, BB * N2, C2, CAT, C1);
    gemm(bk.w[0], bk.b[0], CAT, BUFA, 256, C1 + C2, M); bnact(BUFA, bk.g[0], bk.be[0], 256, M, 0);
    gemm(bk.w[1], bk.b[1], BUFA, OUT, 256, 256, M);     bnact(OUT,  bk.g[1], bk.be[1], 256, M, 0);
  };

  // ---- pipeline ----
  k_transpose_in<<<dim3(N0 / 256, C0, BB), dim3(256), 0, stream>>>(xin, T0);

  sa(T0,  N0,   T0, 16, 2048, 64, 0.2f, b_sa1, 64, 64, 128,  fps1, nx1, P1, bi1);
  sa(nx1, 2048, P1, 128, 1024, 32, 0.4f, b_sa2, 128, 128, 256, fps2, nx2, P2, bi2);
  sa(nx2, 1024, P2, 256, 512, 16, 0.8f, b_sa3, 128, 128, 256,  fps3, nx3, P3, bi3);
  sa(nx3, 512,  P3, 256, 256, 16, 1.2f, b_sa4, 128, 128, 256,  fps4, nx4, P4, bi4);

  fp(nx3, 512,  nx4, 256,  P3, 256, P4,  256, b_fp4, CAT4, L3P);
  fp(nx2, 1024, nx3, 512,  P2, 256, L3P, 256, b_fp3, CAT3, L2P);
  fp(nx1, 2048, nx2, 1024, P1, 128, L2P, 256, b_fp2, CAT2, L1P);

  long long Mv = (long long)BB * 2048;
  gemm(b_vote.w[0], b_vote.b[0], L1P,  BUFA, 256, 256, Mv); bnact(BUFA, b_vote.g[0], b_vote.be[0], 256, Mv, 1);
  gemm(b_vote.w[1], b_vote.b[1], BUFA, BUFB, 256, 256, Mv); bnact(BUFB, b_vote.g[1], b_vote.be[1], 256, Mv, 1);
  gemm(b_vote.w[2], b_vote.b[2], BUFB, BUFA, 3, 256, Mv);

  k_output<<<dim3((BB * 2048 + 255) / 256), dim3(256), 0, stream>>>(
      BUFA, nx1, fps1, (float*)d_out);
}
